// SwinLSTMCell_51256139710471
// MI455X (gfx1250) — compile-verified, matrix operand order checked by reading, not run on hardware
//
#include <hip/hip_runtime.h>
#include <hip/hip_bf16.h>
#include <math.h>

// ---------------------------------------------------------------------------
// SwinLSTM cell forward for MI455X (gfx1250), bf16 WMMA path.
// B=16, H=W=64, C=128, WS=8, NH=4, HD=32, N=64, D=4 blocks.
// ---------------------------------------------------------------------------

typedef __attribute__((ext_vector_type(16))) __bf16 v16bf;
typedef __attribute__((ext_vector_type(8)))  __bf16 v8bf;
typedef __attribute__((ext_vector_type(8)))  float  v8f;

#define CC   128
#define HW   4096           // 64*64
#define NTOK 65536          // B*H*W
#define NWIN 1024           // B * 64 windows
#define SELEM 8388608       // NTOK * CC

// ---- WMMA fragment loader -------------------------------------------------
// 16-bit A/B fragment layout (CDNA5 ISA 7.12.2): for lane L, half = L/16,
// element j in [0,8): k = half*8 + j ; element j in [8,16): k = 16 + half*8 + (j-8).
// `row` points at 32 contiguous bf16 K-values (16B aligned at half*8 offsets).
__device__ __forceinline__ v16bf load_frag(const __bf16* row, int half) {
    v8bf lo = *(const v8bf*)(row + half * 8);
    v8bf hi = *(const v8bf*)(row + 16 + half * 8);
    return __builtin_shufflevector(lo, hi, 0,1,2,3,4,5,6,7,8,9,10,11,12,13,14,15);
}

__device__ __forceinline__ float gelu_exact(float x) {
    return 0.5f * x * (1.0f + erff(x * 0.70710678118654752f));
}

// ---------------------------------------------------------------------------
// Generic GEMM:  Cout(M x Nout) = A(M x K) @ W(K x Nout) + bias [+residual][gelu]
// fp32 in/out, bf16 WMMA compute. Tile: 256x64 per workgroup; 8 waves, each
// wave owns a 32x64 strip (2 A-frags x 4 B-frags = 8 WMMAs per K-step).
// M % 256 == 0, K % 32 == 0, Nout % 64 == 0.
// ---------------------------------------------------------------------------
template<bool GELU, bool RES>
__global__ __launch_bounds__(256) void gemm_bf16_kernel(
    const float* __restrict__ A, const float* __restrict__ W,
    const float* __restrict__ bias, const float* __restrict__ residual,
    float* __restrict__ Cout, int M, int K, int Nout)
{
    __shared__ __align__(16) __bf16 lds_a[256 * 32];   // 16 KB
    __shared__ __align__(16) __bf16 lds_bt[64 * 32];   // 4 KB, n-major / k-contig

    const int m0 = blockIdx.y * 256;
    const int n0 = blockIdx.x * 64;
    const int tid  = threadIdx.x;
    const int wave = tid >> 5;
    const int lane = tid & 31;
    const int half = lane >> 4;
    const int l16  = lane & 15;

    v8f acc[2][4] = {};

    for (int k0 = 0; k0 < K; k0 += 32) {
        // Stage A tile: 256 rows x 32 k (each thread: one full row)
        {
            const float* src = A + (size_t)(m0 + tid) * K + k0;
            __bf16* dst = &lds_a[tid * 32];
            #pragma unroll
            for (int j = 0; j < 32; ++j) dst[j] = (__bf16)src[j];
        }
        // Stage transposed B tile: Bt[n][k] = W[k0+k][n0+n]
        {
            int n  = tid & 63;
            int ks = (tid >> 6) * 8;
            const float* src = W + (size_t)(k0 + ks) * Nout + (n0 + n);
            __bf16* dst = &lds_bt[n * 32 + ks];
            #pragma unroll
            for (int j = 0; j < 8; ++j) dst[j] = (__bf16)src[(size_t)j * Nout];
        }
        __syncthreads();

        // Gather all fragments first (batch the ds_loads), then WMMA back-to-back.
        v16bf afrag[2], bfrag[4];
        #pragma unroll
        for (int r = 0; r < 2; ++r)
            afrag[r] = load_frag(&lds_a[(wave * 32 + r * 16 + l16) * 32], half);
        #pragma unroll
        for (int nt = 0; nt < 4; ++nt)
            bfrag[nt] = load_frag(&lds_bt[(nt * 16 + l16) * 32], half);

        #pragma unroll
        for (int r = 0; r < 2; ++r)
            #pragma unroll
            for (int nt = 0; nt < 4; ++nt)
                acc[r][nt] = __builtin_amdgcn_wmma_f32_16x16x32_bf16(
                    false, afrag[r], false, bfrag[nt], (short)0, acc[r][nt], false, false);
        __syncthreads();
    }

    // Epilogue. D layout: n = l16 within tile, m = half*8 + e.
    #pragma unroll
    for (int r = 0; r < 2; ++r) {
        #pragma unroll
        for (int nt = 0; nt < 4; ++nt) {
            int n = n0 + nt * 16 + l16;
            float bv = bias[n];
            #pragma unroll
            for (int e = 0; e < 8; ++e) {
                int m = m0 + wave * 32 + r * 16 + half * 8 + e;
                float v = acc[r][nt][e] + bv;
                if (GELU) v = gelu_exact(v);
                size_t idx = (size_t)m * Nout + n;
                if (RES) v += residual[idx];
                Cout[idx] = v;
            }
        }
    }
}

// ---------------------------------------------------------------------------
// Windowed attention, one (window, head) per 128-thread workgroup (4 waves).
// qkv: (NWIN, 64, 384)  ->  out: (NWIN, 64, 128)
// ---------------------------------------------------------------------------
__device__ __forceinline__ int region_group(int p) { return (p < 56) ? 0 : (p < 60 ? 1 : 2); }

template<bool SHIFTED>
__global__ __launch_bounds__(128) void attn_kernel(
    const float* __restrict__ qkv, const float* __restrict__ rpb /* (225,4) */,
    float* __restrict__ out)
{
    const int win  = blockIdx.x;
    const int head = blockIdx.y;

    __shared__ __align__(16) __bf16 q_lds[64 * 32];   // scaled Q, A-layout
    __shared__ __align__(16) __bf16 kt_lds[64 * 32];  // K rows  == Bt for Q@K^T
    __shared__ __align__(16) __bf16 vt_lds[32 * 64];  // Vt[n][k] = V[k][n]
    __shared__ float  s_lds[64 * 64];                 // scores / exp scratch
    __shared__ __align__(16) __bf16 p_lds[64 * 64];   // softmax, A-layout

    const int tid  = threadIdx.x;
    const int wave = tid >> 5;
    const int lane = tid & 31;
    const int half = lane >> 4;
    const int l16  = lane & 15;
    const float scale = 0.17677669529663689f;  // 32^-0.5

    // Stage Q,K,V (each thread: one token, half the head-dim)
    {
        int i  = tid >> 1;
        int cs = (tid & 1) * 16;
        const float* base = qkv + ((size_t)win * 64 + i) * 384 + head * 32;
        #pragma unroll
        for (int j = 0; j < 16; ++j) {
            int c = cs + j;
            q_lds[i * 32 + c]  = (__bf16)(base[c] * scale);
            kt_lds[i * 32 + c] = (__bf16)(base[128 + c]);
            vt_lds[c * 64 + i] = (__bf16)(base[256 + c]);
        }
    }
    __syncthreads();

    // S = (Q*scale) @ K^T : each wave a 16x64 strip, single K=32 step
    {
        v16bf afrag = load_frag(&q_lds[(wave * 16 + l16) * 32], half);
        v16bf bfrag[4];
        #pragma unroll
        for (int nt = 0; nt < 4; ++nt)
            bfrag[nt] = load_frag(&kt_lds[(nt * 16 + l16) * 32], half);
        v8f sacc[4] = {};
        #pragma unroll
        for (int nt = 0; nt < 4; ++nt)
            sacc[nt] = __builtin_amdgcn_wmma_f32_16x16x32_bf16(
                false, afrag, false, bfrag[nt], (short)0, sacc[nt], false, false);

        const int wi = win & 63, wh = wi >> 3, ww = wi & 7;
        #pragma unroll
        for (int nt = 0; nt < 4; ++nt) {
            int j = nt * 16 + l16;
            int rj = j >> 3, cj = j & 7;
            int gj = 0;
            if (SHIFTED)
                gj = region_group(wh * 8 + rj) * 3 + region_group(ww * 8 + cj);
            #pragma unroll
            for (int e = 0; e < 8; ++e) {
                int i = wave * 16 + half * 8 + e;
                int ri = i >> 3, ci = i & 7;
                int rel = (ri - rj + 7) * 15 + (ci - cj + 7);
                float v = sacc[nt][e] + rpb[rel * 4 + head];
                if (SHIFTED) {
                    int gi = region_group(wh * 8 + ri) * 3 + region_group(ww * 8 + ci);
                    v += (gi != gj) ? -100.0f : 0.0f;   // branchless cndmask
                }
                s_lds[i * 64 + j] = v;
            }
        }
    }
    __syncthreads();

    // Row softmax (threads 0..63, one row each), no private arrays:
    // pass1 max, pass2 exp (stored back into s_lds), pass3 normalize -> bf16.
    if (tid < 64) {
        float* row = &s_lds[tid * 64];
        float mx = row[0];
        for (int j = 1; j < 64; ++j) mx = fmaxf(mx, row[j]);
        float sum = 0.0f;
        for (int j = 0; j < 64; ++j) { float ev = expf(row[j] - mx); row[j] = ev; sum += ev; }
        float inv = 1.0f / sum;
        for (int j = 0; j < 64; ++j) p_lds[tid * 64 + j] = (__bf16)(row[j] * inv);
    }
    __syncthreads();

    // P = softmax @ V : 16x32 strip per wave, K=64 -> two K-steps
    {
        v8f pacc[2] = {};
        #pragma unroll
        for (int ks = 0; ks < 2; ++ks) {
            v16bf afrag = load_frag(&p_lds[(wave * 16 + l16) * 64 + ks * 32], half);
            v16bf bfrag[2];
            #pragma unroll
            for (int nt = 0; nt < 2; ++nt)
                bfrag[nt] = load_frag(&vt_lds[(nt * 16 + l16) * 64 + ks * 32], half);
            #pragma unroll
            for (int nt = 0; nt < 2; ++nt)
                pacc[nt] = __builtin_amdgcn_wmma_f32_16x16x32_bf16(
                    false, afrag, false, bfrag[nt], (short)0, pacc[nt], false, false);
        }
        #pragma unroll
        for (int nt = 0; nt < 2; ++nt) {
            int n = nt * 16 + l16;
            #pragma unroll
            for (int e = 0; e < 8; ++e) {
                int i = wave * 16 + half * 8 + e;
                out[((size_t)win * 64 + i) * 128 + head * 32 + n] = pacc[nt][e];
            }
        }
    }
}

// ---------------------------------------------------------------------------
// LayerNorm helpers
// ---------------------------------------------------------------------------
__device__ __forceinline__ float block_reduce_sum128(float v, float* sbuf) {
    int t = threadIdx.x;
    sbuf[t] = v; __syncthreads();
    for (int s = 64; s > 0; s >>= 1) { if (t < s) sbuf[t] += sbuf[t + s]; __syncthreads(); }
    float r = sbuf[0]; __syncthreads();
    return r;
}

// LN (+ optional gated concat with ln(hx)) with window-partition + cyclic shift.
// grid.x = NTOK windowed tokens, 128 threads = channels.
// out: (NWIN, 64, Cin) with Cin = hx ? 256 : 128
__global__ __launch_bounds__(128) void ln_window_kernel(
    const float* __restrict__ x, const float* __restrict__ hx,
    const float* __restrict__ w, const float* __restrict__ b,
    float* __restrict__ out, int shift)
{
    __shared__ float sbuf[128];
    int wt   = blockIdx.x;
    int widx = wt >> 6;
    int i    = wt & 63;
    int bimg = widx >> 6;
    int wi   = widx & 63;
    int ph = (((wi >> 3) * 8) + (i >> 3) + shift) & 63;
    int pw = (((wi & 7) * 8) + (i & 7) + shift) & 63;
    size_t tok = (size_t)bimg * HW + ph * 64 + pw;
    int t = threadIdx.x;
    int Cin = hx ? 256 : 128;
    float* dst = out + (size_t)wt * Cin;

    {
        float v = x[tok * CC + t];
        float mean = block_reduce_sum128(v, sbuf) * (1.0f / 128.0f);
        float d = v - mean;
        float var = block_reduce_sum128(d * d, sbuf) * (1.0f / 128.0f);
        dst[t] = d * rsqrtf(var + 1e-5f) * w[t] + b[t];
    }
    if (hx) {
        float v = hx[tok * CC + t];
        float mean = block_reduce_sum128(v, sbuf) * (1.0f / 128.0f);
        float d = v - mean;
        float var = block_reduce_sum128(d * d, sbuf) * (1.0f / 128.0f);
        dst[128 + t] = d * rsqrtf(var + 1e-5f) * w[t] + b[t];
    }
}

// Plain token-major LN (for norm2). grid.x = NTOK, 128 threads.
__global__ __launch_bounds__(128) void ln_flat_kernel(
    const float* __restrict__ x, const float* __restrict__ w,
    const float* __restrict__ b, float* __restrict__ out)
{
    __shared__ float sbuf[128];
    size_t tok = blockIdx.x;
    int t = threadIdx.x;
    float v = x[tok * CC + t];
    float mean = block_reduce_sum128(v, sbuf) * (1.0f / 128.0f);
    float d = v - mean;
    float var = block_reduce_sum128(d * d, sbuf) * (1.0f / 128.0f);
    out[tok * CC + t] = d * rsqrtf(var + 1e-5f) * w[t] + b[t];
}

// Window-reverse + un-shift + residual: x[idx] += P[windowed(idx)]
__global__ __launch_bounds__(256) void scatter_residual_kernel(
    float* __restrict__ x, const float* __restrict__ P, int shift)
{
    size_t idx = (size_t)blockIdx.x * 256 + threadIdx.x;
    int c  = idx & 127;
    size_t tk = idx >> 7;
    int bimg = (int)(tk >> 12);
    int hw = (int)(tk & 4095);
    int hs = ((hw >> 6) - shift) & 63;
    int ws = ((hw & 63) - shift) & 63;
    int widx = bimg * 64 + (hs >> 3) * 8 + (ws >> 3);
    int i = (hs & 7) * 8 + (ws & 7);
    x[idx] += P[((size_t)widx * 64 + i) * 128 + c];
}

// Final LSTM gating: Ft=sigmoid(o); Ct=Ft*(cx+tanh(o)); Ht=Ft*tanh(Ct)
// d_out = [Ht | Ht | Ct]
__global__ __launch_bounds__(256) void lstm_final_kernel(
    const float* __restrict__ o, const float* __restrict__ cx, float* __restrict__ out)
{
    size_t idx = (size_t)blockIdx.x * 256 + threadIdx.x;
    float ov = o[idx];
    float F  = 1.0f / (1.0f + expf(-ov));
    float Ct = F * (cx[idx] + tanhf(ov));
    float Ht = F * tanhf(Ct);
    out[idx] = Ht;
    out[SELEM + idx] = Ht;
    out[2 * (size_t)SELEM + idx] = Ct;
}

// ---------------------------------------------------------------------------
// Host orchestration
// ---------------------------------------------------------------------------
static inline void launch_gemm(const float* A, const float* W, const float* bias,
                               float* C, int M, int K, int N, hipStream_t s)
{
    dim3 grid(N / 64, M / 256);
    gemm_bf16_kernel<false, false><<<grid, 256, 0, s>>>(A, W, bias, nullptr, C, M, K, N);
}
static inline void launch_gemm_gelu(const float* A, const float* W, const float* bias,
                                    float* C, int M, int K, int N, hipStream_t s)
{
    dim3 grid(N / 64, M / 256);
    gemm_bf16_kernel<true, false><<<grid, 256, 0, s>>>(A, W, bias, nullptr, C, M, K, N);
}
static inline void launch_gemm_res(const float* A, const float* W, const float* bias,
                                   const float* res, float* C, int M, int K, int N,
                                   hipStream_t s)
{
    dim3 grid(N / 64, M / 256);
    gemm_bf16_kernel<false, true><<<grid, 256, 0, s>>>(A, W, bias, res, C, M, K, N);
}

extern "C" void kernel_launch(void* const* d_in, const int* in_sizes, int n_in,
                              void* d_out, int out_size, void* d_ws, size_t ws_size,
                              hipStream_t stream) {
    const float* xt      = (const float*)d_in[0];
    const float* hx      = (const float*)d_in[1];
    const float* cx      = (const float*)d_in[2];
    const float* norm1_w = (const float*)d_in[3];
    const float* norm1_b = (const float*)d_in[4];
    const float* red_w   = (const float*)d_in[5];
    const float* red_b   = (const float*)d_in[6];
    const float* qkv_w   = (const float*)d_in[7];
    const float* qkv_b   = (const float*)d_in[8];
    const float* rpb     = (const float*)d_in[9];
    const float* proj_w  = (const float*)d_in[10];
    const float* proj_b  = (const float*)d_in[11];
    const float* norm2_w = (const float*)d_in[12];
    const float* norm2_b = (const float*)d_in[13];
    const float* fc1_w   = (const float*)d_in[14];
    const float* fc1_b   = (const float*)d_in[15];
    const float* fc2_w   = (const float*)d_in[16];
    const float* fc2_b   = (const float*)d_in[17];

    const size_t S = SELEM;
    float* buf_x = (float*)d_ws;          // S     : current activations (token-major)
    float* buf_a = buf_x + S;             // 2S    : windowed LN / ln2 output
    float* buf_c = buf_a + 2 * S;         // S     : reduced-X / proj output
    float* buf_b = buf_c + S;             // 4S    : qkv / MLP hidden

    hipMemcpyAsync(buf_x, xt, S * sizeof(float), hipMemcpyDeviceToDevice, stream);

    for (int i = 0; i < 4; ++i) {
        int shift = (i & 1) ? 4 : 0;
        const float* gate = (i == 0) ? hx : (i == 2 ? xt : nullptr);

        // 1) LN (+gated concat) into windowed layout
        ln_window_kernel<<<NTOK, 128, 0, stream>>>(
            buf_x, gate, norm1_w + i * CC, norm1_b + i * CC, buf_a, shift);

        // 2) optional reduction GEMM (2C -> C)
        const float* X1 = buf_a;
        if (gate) {
            launch_gemm(buf_a, red_w + (size_t)i * 2 * CC * CC, red_b + i * CC,
                        buf_c, NTOK, 2 * CC, CC, stream);
            X1 = buf_c;
        }

        // 3) QKV GEMM (C -> 3C)
        launch_gemm(X1, qkv_w + (size_t)i * CC * 3 * CC, qkv_b + i * 3 * CC,
                    buf_b, NTOK, CC, 3 * CC, stream);

        // 4) windowed multi-head attention
        if (shift > 0)
            attn_kernel<true><<<dim3(NWIN, 4), 128, 0, stream>>>(
                buf_b, rpb + (size_t)i * 225 * 4, buf_a);
        else
            attn_kernel<false><<<dim3(NWIN, 4), 128, 0, stream>>>(
                buf_b, rpb + (size_t)i * 225 * 4, buf_a);

        // 5) output projection
        launch_gemm(buf_a, proj_w + (size_t)i * CC * CC, proj_b + i * CC,
                    buf_c, NTOK, CC, CC, stream);

        // 6) window-reverse + un-shift + residual
        scatter_residual_kernel<<<(unsigned)(S / 256), 256, 0, stream>>>(buf_x, buf_c, shift);

        // 7) LN2
        ln_flat_kernel<<<NTOK, 128, 0, stream>>>(buf_x, norm2_w + i * CC, norm2_b + i * CC, buf_a);

        // 8) MLP fc1 + exact GELU (C -> 4C)
        launch_gemm_gelu(buf_a, fc1_w + (size_t)i * CC * 4 * CC, fc1_b + i * 4 * CC,
                         buf_b, NTOK, CC, 4 * CC, stream);

        // 9) MLP fc2 + residual, in place into buf_x (4C -> C)
        launch_gemm_res(buf_b, fc2_w + (size_t)i * 4 * CC * CC, fc2_b + i * CC,
                        buf_x, buf_x, NTOK, 4 * CC, CC, stream);
    }

    // Final LSTM gating -> d_out = [Ht | Ht | Ct]
    lstm_final_kernel<<<(unsigned)(S / 256), 256, 0, stream>>>(buf_x, cx, (float*)d_out);
}
